// StockXLSTMTS_82867099009372
// MI455X (gfx1250) — matrix operand order repaired and checked by reading, not compiled, not run on hardware
//
#include <hip/hip_runtime.h>

#define BSZ   4096
#define TSEQ  30
#define HD    512
#define GD    (5 * HD)          // 2560 gate columns
#define NHEAD 4
#define DH    128
#define LATD  128
#define OUTD  14
#define EPSF  1e-5f

typedef __attribute__((ext_vector_type(16))) __bf16 bf16x16;
typedef __attribute__((ext_vector_type(8)))  float  v8f;
typedef __attribute__((ext_vector_type(4)))  unsigned int u32x4;
typedef unsigned short u16;

__device__ __forceinline__ u16 f2bf(float f) {
    unsigned int u = __float_as_uint(f);
    u += 0x7FFFu + ((u >> 16) & 1u);   // round-to-nearest-even
    return (u16)(u >> 16);
}
__device__ __forceinline__ float sigf(float x) { return 1.0f / (1.0f + __expf(-x)); }

// ---------------------------------------------------------------------------
// f32 -> bf16 conversion
// ---------------------------------------------------------------------------
__global__ void cvt_bf16_kernel(const float* __restrict__ in, u16* __restrict__ out, int n) {
    int i = blockIdx.x * 256 + threadIdx.x;
    if (i < n) out[i] = f2bf(in[i]);
}

// ---------------------------------------------------------------------------
// WMMA bf16 GEMM:  C[M,N] = A[M,K] * W[N,K]^T (+ bias[N])
// A has two row-major sources: k < K0 reads A0 (stride lda0), else A1 (lda1).
// Each wave: 16(M) x 64(N) tile, block = 8 waves -> 16 x 512.
// grid = (M/16, N/512), block = 256. Requires K % 32 == 0, K0 % 32 == 0.
// ---------------------------------------------------------------------------
union BFrag { bf16x16 v; u32x4 q[2]; };

__global__ __launch_bounds__(256) void gemm_wmma_bf16(
    const u16* __restrict__ A0, int lda0, int K0,
    const u16* __restrict__ A1, int lda1,
    const u16* __restrict__ W,  int ldw,
    const float* __restrict__ bias,
    float* __restrict__ C, int ldc, int K)
{
    const int lane  = threadIdx.x & 31;
    const int wave  = threadIdx.x >> 5;
    const int mTile = blockIdx.x * 16;
    const int nBase = blockIdx.y * 512 + wave * 64;

    const int am = lane & 15;              // A row within tile (ISA A layout)
    const int ak = (lane >> 4) * 16;       // A K-offset for this half-wave
    const int bn = lane & 15;              // B column (N) within tile
    const int bk = (lane >> 4) * 16;       // B K-offset for this half-wave

    const size_t arow = (size_t)(mTile + am);

    v8f acc0 = {}, acc1 = {}, acc2 = {}, acc3 = {};

    for (int k0 = 0; k0 < K; k0 += 32) {
        const u16* aptr = (k0 < K0) ? (A0 + arow * (size_t)lda0 + k0)
                                    : (A1 + arow * (size_t)lda1 + (k0 - K0));
        BFrag af;
        af.q[0] = *(const u32x4*)(aptr + ak);        // K = koff .. koff+7
        af.q[1] = *(const u32x4*)(aptr + ak + 8);    // K = koff+8 .. koff+15

        const u16* w0 = W + (size_t)(nBase + bn) * ldw + k0 + bk;
        __builtin_prefetch(w0 + 32, 0, 0);           // next K-chunk of this row

        BFrag b0, b1, b2, b3;
        b0.q[0] = *(const u32x4*)(w0);
        b0.q[1] = *(const u32x4*)(w0 + 8);
        b1.q[0] = *(const u32x4*)(w0 + 16 * (size_t)ldw);
        b1.q[1] = *(const u32x4*)(w0 + 16 * (size_t)ldw + 8);
        b2.q[0] = *(const u32x4*)(w0 + 32 * (size_t)ldw);
        b2.q[1] = *(const u32x4*)(w0 + 32 * (size_t)ldw + 8);
        b3.q[0] = *(const u32x4*)(w0 + 48 * (size_t)ldw);
        b3.q[1] = *(const u32x4*)(w0 + 48 * (size_t)ldw + 8);

        acc0 = __builtin_amdgcn_wmma_f32_16x16x32_bf16(false, af.v, false, b0.v, (short)0, acc0, false, false);
        acc1 = __builtin_amdgcn_wmma_f32_16x16x32_bf16(false, af.v, false, b1.v, (short)0, acc1, false, false);
        acc2 = __builtin_amdgcn_wmma_f32_16x16x32_bf16(false, af.v, false, b2.v, (short)0, acc2, false, false);
        acc3 = __builtin_amdgcn_wmma_f32_16x16x32_bf16(false, af.v, false, b3.v, (short)0, acc3, false, false);
    }

    // C/D layout: VGPR r holds M = r (lanes 0-15) and M = r+8 (lanes 16-31); N = lane&15
    const int cn = lane & 15;
    const int mo = (lane >> 4) * 8;
    #pragma unroll
    for (int r = 0; r < 8; ++r) {
        float* crow = C + (size_t)(mTile + mo + r) * ldc;
        int n0 = nBase + cn;
        float bsum0 = bias ? bias[n0]      : 0.f;
        float bsum1 = bias ? bias[n0 + 16] : 0.f;
        float bsum2 = bias ? bias[n0 + 32] : 0.f;
        float bsum3 = bias ? bias[n0 + 48] : 0.f;
        crow[n0]      = acc0[r] + bsum0;
        crow[n0 + 16] = acc1[r] + bsum1;
        crow[n0 + 32] = acc2[r] + bsum2;
        crow[n0 + 48] = acc3[r] + bsum3;
    }
}

// ---------------------------------------------------------------------------
// Front-end: conv1d(9->24,k=3,pad=1)+ReLU+BN+maxpool2 ; comp MLP(8->20,ReLU);
// concat(44) -> proj 512 -> bf16.  grid = (B, 30), block = 256.
// ---------------------------------------------------------------------------
__global__ __launch_bounds__(256) void frontend_kernel(
    const float* __restrict__ x,        // (B,60,17)
    const float* __restrict__ c1w, const float* __restrict__ c1b,
    const float* __restrict__ bg,  const float* __restrict__ bb,
    const float* __restrict__ brm, const float* __restrict__ brv,
    const float* __restrict__ cw,  const float* __restrict__ cb,
    const float* __restrict__ pw,  const float* __restrict__ pb,
    u16* __restrict__ combined)         // (B,30,512) bf16
{
    const int b = blockIdx.x;
    const int t = blockIdx.y;
    const int tid = threadIdx.x;
    __shared__ float feat[44];

    if (tid < 24) {
        const int o = tid;
        float mx = -1e30f;
        #pragma unroll
        for (int pp = 0; pp < 2; ++pp) {
            int p = 2 * t + pp;
            float acc = c1b[o];
            #pragma unroll
            for (int i = 0; i < 9; ++i) {
                #pragma unroll
                for (int kk = 0; kk < 3; ++kk) {
                    int pos = p + kk - 1;
                    if (pos >= 0 && pos < 60)
                        acc += x[((size_t)b * 60 + pos) * 17 + i] * c1w[(o * 9 + i) * 3 + kk];
                }
            }
            acc = fmaxf(acc, 0.f);
            acc = (acc - brm[o]) * rsqrtf(brv[o] + EPSF) * bg[o] + bb[o];
            mx = fmaxf(mx, acc);
        }
        feat[o] = mx;
    } else if (tid >= 32 && tid < 52) {
        const int j = tid - 32;
        float acc = cb[j];
        #pragma unroll
        for (int e = 0; e < 8; ++e)
            acc += x[((size_t)b * 60 + t) * 17 + 9 + e] * cw[j * 8 + e];
        feat[24 + j] = fmaxf(acc, 0.f);
    }
    __syncthreads();

    for (int h = tid; h < HD; h += 256) {
        float acc = pb[h];
        const float* wr = pw + (size_t)h * 44;
        #pragma unroll
        for (int j = 0; j < 44; ++j) acc += feat[j] * wr[j];
        combined[((size_t)b * TSEQ + t) * HD + h] = f2bf(acc);
    }
}

// ---------------------------------------------------------------------------
// xLSTM cell elementwise: gates -> cell update (retention + mix gate) ->
// expLN+sigmoid -> mul tanh(c) -> LN -> h (bf16).  grid = B, block = 256.
// ---------------------------------------------------------------------------
__global__ __launch_bounds__(256) void xlstm_cell(
    const float* __restrict__ G,        // (B, 2560)
    float* __restrict__ c,              // (B, 512)
    u16* __restrict__ h_out,            // (B, 512) bf16
    u16* __restrict__ seq_out,          // nullable, row stride seq_stride
    int seq_stride,
    const float* __restrict__ ret,
    const float* __restrict__ eg, const float* __restrict__ eb,
    const float* __restrict__ lg, const float* __restrict__ lb)
{
    __shared__ float s1[256], s2[256];
    const int b = blockIdx.x, tid = threadIdx.x;
    const float* Gb = G + (size_t)b * GD;

    float oval[2], tval[2], hval[2];
    #pragma unroll
    for (int e = 0; e < 2; ++e) {
        const int h = tid + e * 256;
        float gf = sigf(Gb[0 * HD + h]);
        float gi = sigf(Gb[1 * HD + h]);
        float go = sigf(Gb[2 * HD + h]);
        float gc = tanhf(Gb[3 * HD + h]);
        float gm = sigf(Gb[4 * HD + h]);
        float cold = c[(size_t)b * HD + h];
        float r = ret[h];
        float cn = gf * cold + gi * gc;
        cn = cn * r + (1.f - r) * cold;
        cn = gm * cn + (1.f - gm) * cold;
        c[(size_t)b * HD + h] = cn;
        oval[e] = go;
        tval[e] = tanhf(cn);
    }
    // LN over o
    s1[tid] = oval[0] + oval[1];
    s2[tid] = oval[0] * oval[0] + oval[1] * oval[1];
    __syncthreads();
    for (int off = 128; off > 0; off >>= 1) {
        if (tid < off) { s1[tid] += s1[tid + off]; s2[tid] += s2[tid + off]; }
        __syncthreads();
    }
    float mean = s1[0] * (1.f / HD);
    float var  = s2[0] * (1.f / HD) - mean * mean;
    float rstd = rsqrtf(var + EPSF);
    __syncthreads();
    #pragma unroll
    for (int e = 0; e < 2; ++e) {
        const int h = tid + e * 256;
        float oexp = sigf((oval[e] - mean) * rstd * eg[h] + eb[h]);
        hval[e] = oexp * tval[e];
    }
    // LN over h
    s1[tid] = hval[0] + hval[1];
    s2[tid] = hval[0] * hval[0] + hval[1] * hval[1];
    __syncthreads();
    for (int off = 128; off > 0; off >>= 1) {
        if (tid < off) { s1[tid] += s1[tid + off]; s2[tid] += s2[tid + off]; }
        __syncthreads();
    }
    mean = s1[0] * (1.f / HD);
    var  = s2[0] * (1.f / HD) - mean * mean;
    rstd = rsqrtf(var + EPSF);
    #pragma unroll
    for (int e = 0; e < 2; ++e) {
        const int h = tid + e * 256;
        float hn = (hval[e] - mean) * rstd * lg[h] + lb[h];
        u16 bv = f2bf(hn);
        h_out[(size_t)b * HD + h] = bv;
        if (seq_out) seq_out[(size_t)b * seq_stride + h] = bv;
    }
}

// ---------------------------------------------------------------------------
// Attention (last query only). grid = B*NH, block = 128 (= DH).
// kv layout: row (b*30+t) of 1024 floats: [k(head,128) | v(head,128)].
// ---------------------------------------------------------------------------
__global__ __launch_bounds__(128) void attn_last_kernel(
    const float* __restrict__ qlast,    // (B, 512)
    const float* __restrict__ kv,       // (B*30, 1024)
    float* __restrict__ attn_f,         // (B, 512)
    u16* __restrict__ attn_b)           // (B, 512) bf16
{
    const int bh = blockIdx.x;
    const int b  = bh >> 2;
    const int hd = bh & 3;
    const int d  = threadIdx.x;
    __shared__ float qs[DH];
    __shared__ float sc[32];

    qs[d] = qlast[(size_t)b * HD + hd * DH + d];
    __syncthreads();

    if (d < TSEQ) {
        const float* krow = kv + ((size_t)b * TSEQ + d) * 1024 + hd * DH;
        float s = 0.f;
        #pragma unroll 4
        for (int e = 0; e < DH; ++e) s += qs[e] * krow[e];
        sc[d] = s * 0.08838834764831845f;   // 1/sqrt(128)
    }
    __syncthreads();

    float mx = -1e30f;
    for (int j = 0; j < TSEQ; ++j) mx = fmaxf(mx, sc[j]);
    float sum = 0.f;
    for (int j = 0; j < TSEQ; ++j) sum += __expf(sc[j] - mx);
    float inv = 1.f / sum;

    float out = 0.f;
    for (int j = 0; j < TSEQ; ++j) {
        float pj = __expf(sc[j] - mx) * inv;
        out += pj * kv[((size_t)b * TSEQ + j) * 1024 + 512 + hd * DH + d];
    }
    attn_f[(size_t)b * HD + hd * DH + d] = out;
    attn_b[(size_t)b * HD + hd * DH + d] = f2bf(out);
}

// ---------------------------------------------------------------------------
// Head: bneck(512->128) + BN + out(128->14). grid = B, block = 128.
// ---------------------------------------------------------------------------
__global__ __launch_bounds__(128) void head_kernel(
    const float* __restrict__ fin,      // (B, 512)
    const float* __restrict__ bw, const float* __restrict__ bbias,
    const float* __restrict__ g2, const float* __restrict__ b2,
    const float* __restrict__ rm2, const float* __restrict__ rv2,
    const float* __restrict__ ow, const float* __restrict__ ob,
    float* __restrict__ out)            // (B, 14)
{
    const int b = blockIdx.x, n = threadIdx.x;
    __shared__ float lat[LATD];
    const float* fr = fin + (size_t)b * HD;
    float s = bbias[n];
    const float* wr = bw + (size_t)n * HD;
    #pragma unroll 4
    for (int k = 0; k < HD; ++k) s += fr[k] * wr[k];
    s = (s - rm2[n]) * rsqrtf(rv2[n] + EPSF) * g2[n] + b2[n];
    lat[n] = s;
    __syncthreads();
    if (n < OUTD) {
        float o = ob[n];
        #pragma unroll 4
        for (int e = 0; e < LATD; ++e) o += lat[e] * ow[n * LATD + e];
        out[(size_t)b * OUTD + n] = o;
    }
}

// ---------------------------------------------------------------------------
// Host orchestration
// ---------------------------------------------------------------------------
extern "C" void kernel_launch(void* const* d_in, const int* in_sizes, int n_in,
                              void* d_out, int out_size, void* d_ws, size_t ws_size,
                              hipStream_t stream) {
    (void)in_sizes; (void)n_in; (void)out_size; (void)ws_size;
    const float* x        = (const float*)d_in[0];
    const float* conv1_w  = (const float*)d_in[1];
    const float* conv1_b  = (const float*)d_in[2];
    const float* bn1_g    = (const float*)d_in[3];
    const float* bn1_b    = (const float*)d_in[4];
    const float* bn1_rm   = (const float*)d_in[5];
    const float* bn1_rv   = (const float*)d_in[6];
    const float* comp_w   = (const float*)d_in[7];
    const float* comp_b   = (const float*)d_in[8];
    const float* proj_w   = (const float*)d_in[9];
    const float* proj_b   = (const float*)d_in[10];
    const float* gates_w  = (const float*)d_in[11];
    const float* gates_b  = (const float*)d_in[12];
    const float* retention= (const float*)d_in[13];
    const float* expln_g  = (const float*)d_in[14];
    const float* expln_b  = (const float*)d_in[15];
    const float* ln_g     = (const float*)d_in[16];
    const float* ln_b     = (const float*)d_in[17];
    const float* attn_in_w= (const float*)d_in[18];
    const float* attn_in_b= (const float*)d_in[19];
    const float* attn_out_w=(const float*)d_in[20];
    const float* attn_out_b=(const float*)d_in[21];
    const float* bneck_w  = (const float*)d_in[22];
    const float* bneck_b  = (const float*)d_in[23];
    const float* bn2_g    = (const float*)d_in[24];
    const float* bn2_b    = (const float*)d_in[25];
    const float* bn2_rm   = (const float*)d_in[26];
    const float* bn2_rv   = (const float*)d_in[27];
    const float* out_w    = (const float*)d_in[28];
    const float* out_b    = (const float*)d_in[29];
    float* out = (float*)d_out;

    char* ws = (char*)d_ws;
    auto alloc = [&](size_t bytes) -> void* {
        void* p = (void*)ws;
        ws += (bytes + 255) & ~(size_t)255;
        return p;
    };

    u16*   combined = (u16*)  alloc((size_t)BSZ * TSEQ * HD * 2);
    u16*   seqb     = (u16*)  alloc((size_t)BSZ * TSEQ * HD * 2);
    float* G        = (float*)alloc((size_t)BSZ * GD * 4);
    u16*   h0       = (u16*)  alloc((size_t)BSZ * HD * 2);
    u16*   h1       = (u16*)  alloc((size_t)BSZ * HD * 2);
    float* c0       = (float*)alloc((size_t)BSZ * HD * 4);
    float* c1       = (float*)alloc((size_t)BSZ * HD * 4);
    float* kv       = (float*)alloc((size_t)BSZ * TSEQ * 1024 * 4);
    float* qlast    = (float*)alloc((size_t)BSZ * HD * 4);
    float* attn_f   = (float*)alloc((size_t)BSZ * HD * 4);
    u16*   attn_bf  = (u16*)  alloc((size_t)BSZ * HD * 2);
    float* finalb   = (float*)alloc((size_t)BSZ * HD * 4);
    u16*   gwb      = (u16*)  alloc((size_t)2 * GD * 1024 * 2);
    u16*   aiwb     = (u16*)  alloc((size_t)1536 * 512 * 2);
    u16*   aowb     = (u16*)  alloc((size_t)512 * 512 * 2);

    // weight conversion f32 -> bf16
    {
        int n1 = 2 * GD * 1024;
        cvt_bf16_kernel<<<(n1 + 255) / 256, 256, 0, stream>>>(gates_w, gwb, n1);
        int n2 = 1536 * 512;
        cvt_bf16_kernel<<<(n2 + 255) / 256, 256, 0, stream>>>(attn_in_w, aiwb, n2);
        int n3 = 512 * 512;
        cvt_bf16_kernel<<<(n3 + 255) / 256, 256, 0, stream>>>(attn_out_w, aowb, n3);
    }

    // zero-init recurrent state (graph-capture safe)
    hipMemsetAsync(h0, 0, (size_t)BSZ * HD * 2, stream);
    hipMemsetAsync(h1, 0, (size_t)BSZ * HD * 2, stream);
    hipMemsetAsync(c0, 0, (size_t)BSZ * HD * 4, stream);
    hipMemsetAsync(c1, 0, (size_t)BSZ * HD * 4, stream);

    // front end
    frontend_kernel<<<dim3(BSZ, TSEQ), 256, 0, stream>>>(
        x, conv1_w, conv1_b, bn1_g, bn1_b, bn1_rm, bn1_rv,
        comp_w, comp_b, proj_w, proj_b, combined);

    auto gemm = [&](const u16* A0, int lda0, int K0, const u16* A1, int lda1,
                    const u16* W, int ldw, const float* bias,
                    float* C, int ldc, int M, int N, int K) {
        dim3 grid(M / 16, N / 512);
        gemm_wmma_bf16<<<grid, 256, 0, stream>>>(A0, lda0, K0, A1, lda1, W, ldw, bias, C, ldc, K);
    };

    const size_t L1W = (size_t)GD * 1024;   // layer-1 weight offset (elements)

    // recurrent core: 30 sequential steps, 2 layers
    for (int t = 0; t < TSEQ; ++t) {
        // layer 0: A = [combined[:,t,:] | h0]
        gemm(combined + (size_t)t * HD, TSEQ * HD, HD, h0, HD,
             gwb, 1024, gates_b, G, GD, BSZ, GD, 1024);
        xlstm_cell<<<BSZ, 256, 0, stream>>>(G, c0, h0, nullptr, 0,
             retention, expln_g, expln_b, ln_g, ln_b);
        // layer 1: A = [h0_new | h1]
        gemm(h0, HD, HD, h1, HD,
             gwb + L1W, 1024, gates_b + GD, G, GD, BSZ, GD, 1024);
        xlstm_cell<<<BSZ, 256, 0, stream>>>(G, c1, h1, seqb + (size_t)t * HD, TSEQ * HD,
             retention + HD, expln_g + HD, expln_b + HD, ln_g + HD, ln_b + HD);
    }

    // K/V projection for all timesteps: (B*30, 512) @ (1024, 512)^T
    gemm(seqb, HD, HD, seqb, HD,
         aiwb + (size_t)512 * 512, 512, attn_in_b + 512,
         kv, 1024, BSZ * TSEQ, 1024, HD);

    // Q projection for last timestep only: (B, 512) @ (512, 512)^T
    gemm(seqb + (size_t)(TSEQ - 1) * HD, TSEQ * HD, HD,
         seqb + (size_t)(TSEQ - 1) * HD, TSEQ * HD,
         aiwb, 512, attn_in_b, qlast, HD, BSZ, HD, HD);

    // softmax-attention on last query
    attn_last_kernel<<<BSZ * NHEAD, 128, 0, stream>>>(qlast, kv, attn_f, attn_bf);

    // output projection: (B, 512) @ (512, 512)^T
    gemm(attn_bf, HD, HD, attn_bf, HD, aowb, 512, attn_out_b,
         finalb, HD, BSZ, HD, HD);

    // bottleneck + BN + output head
    head_kernel<<<BSZ, 128, 0, stream>>>(finalb, bneck_w, bneck_b,
         bn2_g, bn2_b, bn2_rm, bn2_rv, out_w, out_b, out);
}
